// GatingNetwork_26087631356433
// MI455X (gfx1250) — matrix operand order repaired and checked by reading, not compiled, not run on hardware
//
#include <hip/hip_runtime.h>
#include <hip/hip_bf16.h>
#include <math.h>

#define NUM_EXPERTS 64
#define D_DIM 2048
#define N_TOKENS 16384
#define NW (NUM_EXPERTS * D_DIM) // 131072 elements per W matrix

typedef __attribute__((ext_vector_type(16))) __bf16 v16bf;
typedef __attribute__((ext_vector_type(8)))  float  v8f;
typedef __attribute__((ext_vector_type(8)))  unsigned int v8u;

// 16-byte int vector in explicit address spaces for the async-copy builtin.
typedef int v4i __attribute__((vector_size(16)));
typedef __attribute__((address_space(1))) v4i as1_v4i;
typedef __attribute__((address_space(3))) v4i as3_v4i;

#if __has_builtin(__builtin_amdgcn_global_load_async_to_lds_b128) && \
    __has_builtin(__builtin_amdgcn_s_wait_asynccnt)
#define USE_ASYNC_STAGE 1
#else
#define USE_ASYNC_STAGE 0
#endif

__device__ __forceinline__ unsigned short f32_to_bf16_rne(float f) {
  unsigned u = __builtin_bit_cast(unsigned, f);
  unsigned r = u + 0x7FFFu + ((u >> 16) & 1u);
  return (unsigned short)(r >> 16);
}
__device__ __forceinline__ float bf16us_to_f32(unsigned short h) {
  unsigned u = ((unsigned)h) << 16;
  return __builtin_bit_cast(float, u);
}

// Split two consecutive f32 into packed bf16 hi-pair / lo-pair (A-operand layout).
// hi = truncate-to-bf16 (top 16 bits), lo = bf16(f - hi_f) (exact subtraction).
// 6 VALU per pair: 2x v_and, 2x v_sub, 2x v_perm_b32. f0 lands in [15:0].
struct HiLo { unsigned hi, lo; };
__device__ __forceinline__ HiLo split_pair(float f0, float f1) {
  unsigned u0 = __builtin_bit_cast(unsigned, f0);
  unsigned u1 = __builtin_bit_cast(unsigned, f1);
  HiLo r;
  r.hi = __builtin_amdgcn_perm(u1, u0, 0x07060302u);
  float h0 = __builtin_bit_cast(float, u0 & 0xFFFF0000u);
  float h1 = __builtin_bit_cast(float, u1 & 0xFFFF0000u);
  float l0 = f0 - h0;
  float l1 = f1 - h1;
  r.lo = __builtin_amdgcn_perm(__builtin_bit_cast(unsigned, l1),
                               __builtin_bit_cast(unsigned, l0), 0x07060302u);
  return r;
}

// 32-byte block for B-operand loads: two b128 LDS loads -> one v16bf.
struct U8x32 { uint4 a, b; };

// ---------------------------------------------------------------------------
// Kernel 1: split Wg / Wnoise into bf16 hi/lo planes in ws; zero importance.
// ws layout (unsigned short): [0]=Wg_hi, [NW]=Wg_lo, [2NW]=Wn_hi, [3NW]=Wn_lo
// ---------------------------------------------------------------------------
__global__ void moe_prep_kernel(const float* __restrict__ Wg,
                                const float* __restrict__ Wn,
                                unsigned short* __restrict__ wsplit,
                                float* __restrict__ importance) {
  int idx = blockIdx.x * blockDim.x + threadIdx.x;
  if (idx < NW) {
    float g = Wg[idx];
    unsigned short gh = f32_to_bf16_rne(g);
    wsplit[idx]          = gh;
    wsplit[NW + idx]     = f32_to_bf16_rne(g - bf16us_to_f32(gh));
    float n = Wn[idx];
    unsigned short nh = f32_to_bf16_rne(n);
    wsplit[2 * NW + idx] = nh;
    wsplit[3 * NW + idx] = f32_to_bf16_rne(n - bf16us_to_f32(nh));
  }
  if (idx < NUM_EXPERTS) importance[idx] = 0.f;
}

// ---------------------------------------------------------------------------
// Kernel 2: fused GEMM (split-bf16 WMMA) + noisy top-2 gating epilogue.
// Grid: 128 blocks x 128 threads. Wave (32 lanes) owns 32 tokens x 64 experts.
// ---------------------------------------------------------------------------
__global__ __launch_bounds__(128) void moe_gate_kernel(
    const float* __restrict__ x,
    const unsigned short* __restrict__ wsplit,
    const float* __restrict__ eps,
    float* __restrict__ wout,
    float* __restrict__ importance) {

  __shared__ union {
    unsigned short w[4][64][72];   // staged W chunk: 4 planes x 64 experts x 64k (+8 pad)
    float lg[4][2][16][68];        // epilogue: [wave][mat][token][expert(+4 pad)]
  } sm;
  __shared__ float impLds[NUM_EXPERTS];

  const int tid  = threadIdx.x;
  const int lane = tid & 31;
  const int wave = tid >> 5;
  const int l15  = lane & 15;
  const int lh   = (lane >> 4) & 1;
  const int rowBase = blockIdx.x * 128 + wave * 32;

  if (tid < NUM_EXPERTS) impLds[tid] = 0.f;

  const v8f vzero = {0.f, 0.f, 0.f, 0.f, 0.f, 0.f, 0.f, 0.f};
  v8f accG[2][4], accN[2][4];
#pragma unroll
  for (int tt = 0; tt < 2; ++tt)
#pragma unroll
    for (int et = 0; et < 4; ++et) { accG[tt][et] = vzero; accN[tt][et] = vzero; }

  for (int kc = 0; kc < D_DIM; kc += 64) {
    __syncthreads();  // previous chunk's LDS reads done (also covers impLds init)
    // Stage W chunk: 4 planes x 64 experts x 64 bf16 = 2048 x 16B.
#if USE_ASYNC_STAGE
#pragma unroll
    for (int it = 0; it < 16; ++it) {
      int i = tid + it * 128;
      int b = i >> 9, rem = i & 511, e = rem >> 3, g = rem & 7;
      __builtin_amdgcn_global_load_async_to_lds_b128(
          (as1_v4i*)(wsplit + (size_t)b * NW + (size_t)e * D_DIM + kc + g * 8),
          (as3_v4i*)(&sm.w[b][e][g * 8]),
          0, 0);
    }
    __builtin_amdgcn_s_wait_asynccnt(0);
#else
#pragma unroll 1
    for (int half = 0; half < 2; ++half) {
#pragma unroll
      for (int it = 0; it < 8; ++it) {
        int i = tid + (half * 8 + it) * 128;
        int b = i >> 9, rem = i & 511, e = rem >> 3, g = rem & 7;
        const uint4 v = *reinterpret_cast<const uint4*>(
            wsplit + (size_t)b * NW + (size_t)e * D_DIM + kc + g * 8);
        *reinterpret_cast<uint4*>(&sm.w[b][e][g * 8]) = v;
      }
    }
#endif
    __syncthreads();

#pragma unroll
    for (int s = 0; s < 2; ++s) {
      const int k0 = kc + s * 32;
      // A operand (x rows) in 16x32 bf16 layout:
      // lanes 0-15 hold row l15, K {0..7,16..23}; lanes 16-31 same rows, K {8..15,24..31}
      v16bf xhi[2], xlo[2];
#pragma unroll
      for (int tt = 0; tt < 2; ++tt) {
        const float* xp = x + (size_t)(rowBase + tt * 16 + l15) * D_DIM + k0 + lh * 8;
        float4 f0 = *reinterpret_cast<const float4*>(xp);
        float4 f1 = *reinterpret_cast<const float4*>(xp + 4);
        float4 f2 = *reinterpret_cast<const float4*>(xp + 16);
        float4 f3 = *reinterpret_cast<const float4*>(xp + 20);
        v8u hu, lu;
        HiLo p;
        p = split_pair(f0.x, f0.y); hu[0] = p.hi; lu[0] = p.lo;
        p = split_pair(f0.z, f0.w); hu[1] = p.hi; lu[1] = p.lo;
        p = split_pair(f1.x, f1.y); hu[2] = p.hi; lu[2] = p.lo;
        p = split_pair(f1.z, f1.w); hu[3] = p.hi; lu[3] = p.lo;
        p = split_pair(f2.x, f2.y); hu[4] = p.hi; lu[4] = p.lo;
        p = split_pair(f2.z, f2.w); hu[5] = p.hi; lu[5] = p.lo;
        p = split_pair(f3.x, f3.y); hu[6] = p.hi; lu[6] = p.lo;
        p = split_pair(f3.z, f3.w); hu[7] = p.hi; lu[7] = p.lo;
        xhi[tt] = __builtin_bit_cast(v16bf, hu);
        xlo[tt] = __builtin_bit_cast(v16bf, lu);
      }
      const int soff = s * 32 + lh * 8;
#pragma unroll
      for (int m = 0; m < 2; ++m) {
#pragma unroll
        for (int et = 0; et < 4; ++et) {
          const int erow = et * 16 + l15;
          // B operand (W^T) 32x16: lanes 0-15 K {0..7,16..23} of col erow, lanes 16-31 K {8..15,24..31}
          U8x32 uh, ul;
          uh.a = *reinterpret_cast<const uint4*>(&sm.w[m * 2][erow][soff]);
          uh.b = *reinterpret_cast<const uint4*>(&sm.w[m * 2][erow][soff + 16]);
          ul.a = *reinterpret_cast<const uint4*>(&sm.w[m * 2 + 1][erow][soff]);
          ul.b = *reinterpret_cast<const uint4*>(&sm.w[m * 2 + 1][erow][soff + 16]);
          v16bf bhi = __builtin_bit_cast(v16bf, uh);
          v16bf blo = __builtin_bit_cast(v16bf, ul);
#pragma unroll
          for (int tt = 0; tt < 2; ++tt) {
            v8f c = m ? accN[tt][et] : accG[tt][et];
            c = __builtin_amdgcn_wmma_f32_16x16x32_bf16(false, xhi[tt], false, bhi,
                                                        (short)0, c, false, false);
            c = __builtin_amdgcn_wmma_f32_16x16x32_bf16(false, xhi[tt], false, blo,
                                                        (short)0, c, false, false);
            c = __builtin_amdgcn_wmma_f32_16x16x32_bf16(false, xlo[tt], false, bhi,
                                                        (short)0, c, false, false);
            if (m) accN[tt][et] = c; else accG[tt][et] = c;
          }
        }
      }
    }
  }

  // ---------------- epilogue: per-token noisy top-2 softmax ----------------
#pragma unroll 1
  for (int tt = 0; tt < 2; ++tt) {
    __syncthreads();  // prior LDS consumers done before overwrite
    // C layout: lane holds expert et*16+l15, tokens M = r + 8*lh (r = VGPR 0..7)
#pragma unroll
    for (int m = 0; m < 2; ++m)
#pragma unroll
      for (int et = 0; et < 4; ++et) {
        v8f c = m ? accN[tt][et] : accG[tt][et];
#pragma unroll
        for (int r = 0; r < 8; ++r)
          sm.lg[wave][m][lh * 8 + r][et * 16 + l15] = c[r];
      }
    __syncthreads();

    const int t = l15;
    const int token = rowBase + tt * 16 + t;
    const float4* g4p = reinterpret_cast<const float4*>(&sm.lg[wave][0][t][lh * 32]);
    const float4* n4p = reinterpret_cast<const float4*>(&sm.lg[wave][1][t][lh * 32]);
    const float4* e4p = reinterpret_cast<const float4*>(eps + (size_t)token * 64 + lh * 32);

    float v1 = -INFINITY, v2 = -INFINITY;
    int i1 = 0, i2 = 0;
#pragma unroll 2
    for (int jq = 0; jq < 8; ++jq) {
      float4 rg = g4p[jq];
      float4 rn = n4p[jq];
      float4 ev = e4p[jq];
      float gg[4] = {rg.x, rg.y, rg.z, rg.w};
      float nn[4] = {rn.x, rn.y, rn.z, rn.w};
      float ee[4] = {ev.x, ev.y, ev.z, ev.w};
#pragma unroll
      for (int c = 0; c < 4; ++c) {
        int e = lh * 32 + jq * 4 + c;
        // softplus; __logf/__expf precision (~1e-7 rel) is ample for gating
        float sp = (nn[c] > 20.f) ? nn[c] : __logf(1.f + __expf(nn[c]));
        float lv = gg[c] + sp * ee[c];
        if (lv > v1)      { v2 = v1; i2 = i1; v1 = lv; i1 = e; }
        else if (lv > v2) { v2 = lv; i2 = e; }
      }
    }
    // merge the two half-lane partial top-2 lists (tie -> lower index, like lax.top_k)
    float ov1 = __shfl_xor(v1, 16, 32);
    float ov2 = __shfl_xor(v2, 16, 32);
    int   oi1 = __shfl_xor(i1, 16, 32);
    int   oi2 = __shfl_xor(i2, 16, 32);
    float t1v, t2v; int t1i, t2i;
    bool aFirst = (v1 > ov1) || (v1 == ov1 && i1 < oi1);
    if (aFirst) {
      t1v = v1; t1i = i1;
      bool sA = (v2 > ov1) || (v2 == ov1 && i2 < oi1);
      t2v = sA ? v2 : ov1; t2i = sA ? i2 : oi1;
    } else {
      t1v = ov1; t1i = oi1;
      bool sB = (ov2 > v1) || (ov2 == v1 && oi2 < i1);
      t2v = sB ? ov2 : v1; t2i = sB ? oi2 : i1;
    }
    float e2 = __expf(t2v - t1v);
    float wA = 1.f / (1.f + e2);   // weight of top-1
    float wB = e2 / (1.f + e2);    // weight of top-2
    if (lh == 0) {
      atomicAdd(&impLds[t1i], wA);
      atomicAdd(&impLds[t2i], wB);
    }
    // sparse weight row: zeros except top-2 entries
    float* orow = wout + (size_t)token * 64 + lh * 32;
#pragma unroll
    for (int g = 0; g < 8; ++g) {
      int base = lh * 32 + g * 4;
      float w0 = (base + 0 == t1i) ? wA : ((base + 0 == t2i) ? wB : 0.f);
      float w1 = (base + 1 == t1i) ? wA : ((base + 1 == t2i) ? wB : 0.f);
      float w2 = (base + 2 == t1i) ? wA : ((base + 2 == t2i) ? wB : 0.f);
      float w3 = (base + 3 == t1i) ? wA : ((base + 3 == t2i) ? wB : 0.f);
      *reinterpret_cast<float4*>(orow + g * 4) = make_float4(w0, w1, w2, w3);
    }
  }

  __syncthreads();
  if (tid < NUM_EXPERTS) atomicAdd(&importance[tid], impLds[tid]);
}

// ---------------------------------------------------------------------------
// Kernel 3: loss = 0.01 * var(importance) / mean(importance)^2
// ---------------------------------------------------------------------------
__global__ void moe_loss_kernel(const float* __restrict__ importance,
                                float* __restrict__ lossOut) {
  __shared__ float ss[64], sq[64];
  int t = threadIdx.x;
  float v = importance[t];
  ss[t] = v; sq[t] = v * v;
  __syncthreads();
  for (int o = 32; o > 0; o >>= 1) {
    if (t < o) { ss[t] += ss[t + o]; sq[t] += sq[t + o]; }
    __syncthreads();
  }
  if (t == 0) {
    float mean = ss[0] * (1.f / 64.f);
    float var  = sq[0] * (1.f / 64.f) - mean * mean;
    lossOut[0] = 0.01f * var / (mean * mean);
  }
}

extern "C" void kernel_launch(void* const* d_in, const int* in_sizes, int n_in,
                              void* d_out, int out_size, void* d_ws, size_t ws_size,
                              hipStream_t stream) {
  const float* x   = (const float*)d_in[0];
  const float* Wg  = (const float*)d_in[1];
  const float* Wn  = (const float*)d_in[2];
  const float* eps = (const float*)d_in[3];
  float* wout = (float*)d_out;

  unsigned short* wsplit = (unsigned short*)d_ws;
  float* importance = (float*)((char*)d_ws + (size_t)4 * NW * sizeof(unsigned short));

  moe_prep_kernel<<<(NW + 255) / 256, 256, 0, stream>>>(Wg, Wn, wsplit, importance);
  moe_gate_kernel<<<N_TOKENS / 128, 128, 0, stream>>>(x, wsplit, eps, wout, importance);
  moe_loss_kernel<<<1, 64, 0, stream>>>(importance, wout + (size_t)N_TOKENS * 64);
}
